// model_reconstruct_60876866454165
// MI455X (gfx1250) — matrix operand order repaired and checked by reading, not compiled, not run on hardware
//
#include <hip/hip_runtime.h>
#include <hip/hip_bf16.h>
#include <math.h>
#include <stdint.h>

// ---------------------------------------------------------------------------
// Problem constants (reference): N=8192 rows, H=512 hidden, tau=0.8
// ---------------------------------------------------------------------------
#define NROWS   8192
#define HDIM    512
#define RBLK    (NROWS / 16)   // 512 tiles of 16 per dimension
#define KSTEPS  (HDIM / 32)    // 16 WMMA K-steps of 32
#define TAU_INV 1.25f          // 1/0.8

typedef __attribute__((ext_vector_type(16))) __bf16       v16bf;
typedef __attribute__((ext_vector_type(8)))  float        v8f;
typedef __attribute__((ext_vector_type(4)))  unsigned int v4u;
typedef __attribute__((ext_vector_type(8)))  int          v8i;
typedef __attribute__((ext_vector_type(4)))  int          v4i;

// Workspace layout (bytes):
//   [0, 8MB)    : z1 normalized, bf16, A-fragment layout
//   [8MB, 16MB) : z2 normalized, bf16, B-fragment layout
//   [16MB, +8)  : float acc[2] = {sum_pos, sum_total}
#define ZBYTES  ((size_t)NROWS * HDIM * 2)
#define ACC_OFF (2 * ZBYTES)

#if __has_builtin(__builtin_amdgcn_tensor_load_to_lds)
#define HAVE_TDM 1
#else
#define HAVE_TDM 0
#endif

// ---------------------------------------------------------------------------
// TDM 2D load: 8 rows of 1024 B at 16 KB stride -> contiguous 8 KB in LDS.
// D# per CDNA5 ISA 8.3/8.4: data_size=2 (4 B), tile 256x8, dim0_stride=4096.
// This toolchain's builtin takes 6 args (g0, g1, g2, g3, extra, cpol).
// ---------------------------------------------------------------------------
#if HAVE_TDM
__device__ __forceinline__ void tdm_load_2d(uint32_t lds_off, const void* gptr)
{
    const uint64_t ga = (uint64_t)(uintptr_t)gptr;
    v4u g0;
    g0[0] = 1u;                                   // count=1 (valid user D#)
    g0[1] = lds_off;                              // lds_addr (bytes)
    g0[2] = (uint32_t)ga;                         // global_addr[31:0]
    g0[3] = (uint32_t)(ga >> 32) | (2u << 30);    // global_addr[56:32] | type=2
    v8i g1;
    g1[0] = (int)(2u << 16);                      // data_size=2 (4B units)
    g1[1] = (int)(4096u << 16);                   // tensor_dim0 = 4096 (lo16)
    g1[2] = (int)(0x8000u << 16);                 // tensor_dim1 = 0x8000 (lo16)
    g1[3] = (int)(256u << 16);                    // tile_dim0 = 256 elems (1 KB)
    g1[4] = 8;                                    // tile_dim1 = 8, tile_dim2 = 0
    g1[5] = 4096;                                 // tensor_dim0_stride = 16 KB
    g1[6] = 0;
    g1[7] = 0;
    const v4i z4 = {0, 0, 0, 0};                  // groups 2/3 unused (2D tile)
    const v8i z8 = {0, 0, 0, 0, 0, 0, 0, 0};
    __builtin_amdgcn_tensor_load_to_lds(g0, g1, z4, z4, z8, 0);
}
#endif

// Stage one K-step (A: 8 KB, B: 8 KB) into LDS buffer `buf`.
__device__ __forceinline__ void stage_step(char* smem, int buf,
                                           const char* Achunk, const char* Bchunk,
                                           int s, int tid, int wave)
{
#if HAVE_TDM
    if (wave == 0) {
        const uint32_t lbase = (uint32_t)(uintptr_t)smem + (uint32_t)buf * 16384u;
        tdm_load_2d(lbase,         Achunk + (size_t)s * 1024);
        tdm_load_2d(lbase + 8192u, Bchunk + (size_t)s * 1024);
    }
#else
    (void)wave;
    #pragma unroll
    for (int r = 0; r < 4; ++r) {                 // 1024 x 16B units / 256 thr
        const int u   = tid + r * 256;
        const int isB = (u >= 512) ? 1 : 0;
        const int v   = u & 511;
        const int t   = v >> 6;                   // row 0..7
        const int w   = v & 63;                   // 16B unit in row
        const char* src = (isB ? Bchunk : Achunk) +
                          (size_t)s * 1024 + (size_t)t * 16384 + (size_t)w * 16;
        char* dst = smem + (size_t)buf * 16384 + (size_t)isB * 8192 +
                    (size_t)t * 1024 + (size_t)w * 16;
        *(uint4*)dst = *(const uint4*)src;
    }
#endif
}

// ---------------------------------------------------------------------------
// Kernel 1: z = elu(x @ W^T + b), row-normalize, pack bf16 WMMA fragments.
// One 256-thread block handles 16 rows (full 16 x 512 output tile).
// Fragment layouts (ISA 7.12.2, 16-bit, wave32):
//   A: lane L holds row M=L%16; slots 0..7 = K base1+0..7, 8..15 = base2+0..7,
//      base1=(L<16?0:8), base2=base1+16.
//   B: lane L holds col N=L%16; same K slot order as A.
// Fragment (row-block r, step s) at element ((r*KSTEPS+s)*32+lane)*16.
// ---------------------------------------------------------------------------
__global__ __launch_bounds__(256)
void proj_norm_pack(const float* __restrict__ X, const float* __restrict__ W,
                    const float* __restrict__ bias, __bf16* __restrict__ out,
                    int layoutB)
{
    __shared__ float zt[16][HDIM + 4];
    __shared__ float norm2[16];
    __shared__ float invn[16];

    const int tid  = threadIdx.x;
    const int lane = tid & 31;
    const int wave = tid >> 5;
    const int rblk = blockIdx.x;
    const int M16  = lane & 15;
    const bool hi  = lane >= 16;
    const int a_b1 = hi ? 8 : 0;

    if (tid < 16) norm2[tid] = 0.0f;

    const float* xrow = X + (size_t)(rblk * 16 + M16) * HDIM;
    for (int nb = wave * 4; nb < wave * 4 + 4; ++nb) {
        const int ncol = nb * 16 + M16;
        const float* wrow = W + (size_t)ncol * HDIM;
        v8f c = {};
        for (int s = 0; s < KSTEPS; ++s) {
            const int k0 = s * 32;
            v8f a0 = *(const v8f*)(xrow + k0 + a_b1);
            v8f a1 = *(const v8f*)(xrow + k0 + a_b1 + 16);
            v8f b0 = *(const v8f*)(wrow + k0 + a_b1);
            v8f b1 = *(const v8f*)(wrow + k0 + a_b1 + 16);
            v16bf af, bf;
            #pragma unroll
            for (int t = 0; t < 8; ++t) {
                af[t]     = (__bf16)a0[t];
                af[t + 8] = (__bf16)a1[t];
                bf[t]     = (__bf16)b0[t];
                bf[t + 8] = (__bf16)b1[t];
            }
            c = __builtin_amdgcn_wmma_f32_16x16x32_bf16(
                    false, af, false, bf, (short)0, c, false, false);
        }
        const float bv = bias[ncol];
        #pragma unroll
        for (int p = 0; p < 8; ++p) {
            const int M = p + (hi ? 8 : 0);
            float v = c[p] + bv;
            v = v > 0.0f ? v : expm1f(v);
            zt[M][nb * 16 + M16] = v;
        }
    }
    __syncthreads();

    {   // row L2 norms: 16 threads per row
        const int row = tid & 15;
        const int seg = tid >> 4;
        float s = 0.0f;
        for (int q = seg; q < HDIM; q += 16) { const float v = zt[row][q]; s += v * v; }
        atomicAdd(&norm2[row], s);
    }
    __syncthreads();
    if (tid < 16) invn[tid] = rsqrtf(norm2[tid]);
    __syncthreads();

    const float sc = invn[M16];
    for (int s = wave * 2; s < wave * 2 + 2; ++s) {
        const int k0 = s * 32;
        v16bf f;
        if (!layoutB) {          // A-fragment layout (z1)
            #pragma unroll
            for (int t = 0; t < 8; ++t) {
                f[t]     = (__bf16)(zt[M16][k0 + a_b1 + t] * sc);
                f[t + 8] = (__bf16)(zt[M16][k0 + a_b1 + 16 + t] * sc);
            }
        } else {                 // B-fragment layout (z2)
            const int kb = k0 + a_b1;
            #pragma unroll
            for (int t = 0; t < 16; ++t)
                f[t] = (__bf16)(zt[M16][kb + (t & 7) + ((t >> 3) << 4)] * sc);
        }
        *(v16bf*)(out + ((size_t)(rblk * KSTEPS + s) * 32 + lane) * 16) = f;
    }
}

// ---------------------------------------------------------------------------
// Kernel 2: fused m = exp((zn1 @ zn2^T)/tau); acc += {m*pos, m*(pos+neg)}.
// Workgroup = 8x8 tiles (128x128 elems); fragments staged once per workgroup
// per K-step via TDM (double buffered, overlapped with WMMA). Each wave
// register-blocks 2 i-tiles x 4 j-tiles (8 accumulators, 8 WMMA per step).
// ---------------------------------------------------------------------------
__global__ __launch_bounds__(256)
void sim_reduce(const __bf16* __restrict__ Abuf, const __bf16* __restrict__ Bbuf,
                const float* __restrict__ pos, const float* __restrict__ neg,
                float* __restrict__ acc)
{
    __shared__ __align__(32) char smem[2 * 16384];   // [buf][A 8KB | B 8KB]

    const int tid  = threadIdx.x;
    const int lane = tid & 31;
    const int wave = tid >> 5;
    const int wi   = wave & 3;          // i-tile pair: {2wi, 2wi+1}
    const int wj   = wave >> 2;         // j-tile quad: {4wj..4wj+3}
    const int Isup = blockIdx.y;        // 0..63
    const int Jsup = blockIdx.x;        // 0..63

    const char* Achunk = (const char*)Abuf + (size_t)Isup * 8 * KSTEPS * 1024;
    const char* Bchunk = (const char*)Bbuf + (size_t)Jsup * 8 * KSTEPS * 1024;

    v8f c0[4] = {}, c1[4] = {};

    stage_step(smem, 0, Achunk, Bchunk, 0, tid, wave);

    for (int s = 0; s < KSTEPS; ++s) {
#if HAVE_TDM
        if (wave == 0) __builtin_amdgcn_s_wait_tensorcnt(0);
#endif
        __syncthreads();                           // step-s data visible
        if (s + 1 < KSTEPS)                        // overlap DMA with compute
            stage_step(smem, (s + 1) & 1, Achunk, Bchunk, s + 1, tid, wave);

        const char* As = smem + (size_t)(s & 1) * 16384;
        const char* Bs = As + 8192;
        const v16bf a0 = *(const v16bf*)(As + ((size_t)(wi * 2 + 0) * 32 + lane) * 32);
        const v16bf a1 = *(const v16bf*)(As + ((size_t)(wi * 2 + 1) * 32 + lane) * 32);
        #pragma unroll
        for (int q = 0; q < 4; ++q) {
            const v16bf b = *(const v16bf*)(Bs + ((size_t)(wj * 4 + q) * 32 + lane) * 32);
            c0[q] = __builtin_amdgcn_wmma_f32_16x16x32_bf16(
                        false, a0, false, b, (short)0, c0[q], false, false);
            c1[q] = __builtin_amdgcn_wmma_f32_16x16x32_bf16(
                        false, a1, false, b, (short)0, c1[q], false, false);
        }
        __syncthreads();                           // reads done before reuse
    }

    // C layout: lanes 0-15 -> M=p, lanes 16-31 -> M=8+p; N = lane%16.
    const int N     = lane & 15;
    const int mbase = (lane >= 16) ? 8 : 0;
    float sp = 0.0f, st = 0.0f;
    #pragma unroll
    for (int ti = 0; ti < 2; ++ti) {
        #pragma unroll
        for (int tj = 0; tj < 4; ++tj) {
            const int    gi0 = (Isup * 8 + wi * 2 + ti) * 16;
            const size_t gj  = (size_t)(Jsup * 8 + wj * 4 + tj) * 16 + N;
            const v8f    cc  = ti ? c1[tj] : c0[tj];
            #pragma unroll
            for (int p = 0; p < 8; ++p) {
                const size_t idx = (size_t)(gi0 + mbase + p) * NROWS + gj;
                const float m  = __expf(cc[p] * TAU_INV);
                const float pv = pos[idx];
                const float nv = neg[idx];
                sp += m * pv;
                st += m * (pv + nv);
            }
        }
    }
    #pragma unroll
    for (int off = 16; off > 0; off >>= 1) {
        sp += __shfl_down(sp, off, 32);
        st += __shfl_down(st, off, 32);
    }
    if (lane == 0) {
        atomicAdd(&acc[0], sp);
        atomicAdd(&acc[1], st);
    }
}

__global__ void init_acc(float* acc) { acc[0] = 0.0f; acc[1] = 0.0f; }

__global__ void finalize(const float* __restrict__ acc, float* __restrict__ out)
{
    out[0] = logf(acc[1]) - logf(acc[0]);          // -log(sum_pos / sum_total)
}

// ---------------------------------------------------------------------------
extern "C" void kernel_launch(void* const* d_in, const int* in_sizes, int n_in,
                              void* d_out, int out_size, void* d_ws, size_t ws_size,
                              hipStream_t stream)
{
    (void)in_sizes; (void)n_in; (void)out_size; (void)ws_size;
    const float* v1  = (const float*)d_in[0];
    const float* v2  = (const float*)d_in[1];
    const float* pos = (const float*)d_in[2];
    const float* neg = (const float*)d_in[3];
    const float* W   = (const float*)d_in[4];
    const float* b   = (const float*)d_in[5];

    __bf16* Abuf = (__bf16*)d_ws;
    __bf16* Bbuf = (__bf16*)((char*)d_ws + ZBYTES);
    float*  acc  = (float*)((char*)d_ws + ACC_OFF);
    float*  out  = (float*)d_out;

    init_acc<<<1, 1, 0, stream>>>(acc);
    proj_norm_pack<<<RBLK, 256, 0, stream>>>(v1, W, b, Abuf, 0);
    proj_norm_pack<<<RBLK, 256, 0, stream>>>(v2, W, b, Bbuf, 1);
    sim_reduce<<<dim3(RBLK / 8, RBLK / 8), 256, 0, stream>>>(Abuf, Bbuf, pos, neg, acc);
    finalize<<<1, 1, 0, stream>>>(acc, out);
}
